// HGRN2Attention_33200097199003
// MI455X (gfx1250) — compile-verified, matrix operand order checked by reading, not run on hardware
//
#include <hip/hip_runtime.h>
#include <hip/hip_bf16.h>

// ---------------------------------------------------------------------------
// HGRN2 attention block for MI455X (gfx1250, wave32, WMMA + TDM).
//   * one-time fp32->bf16 conversion (activations) + transpose-convert (weights)
//   * bf16 WMMA GEMMs (fp32 accum); BOTH tiles staged by Tensor Data Mover,
//     double-buffered (issue i+1, wait tensorcnt<=2, consume i)
//   * GLA scan: register-resident 128x128 state per (b,h)
// ---------------------------------------------------------------------------

typedef __attribute__((ext_vector_type(16))) __bf16 v16bf;
typedef __attribute__((ext_vector_type(8)))  __bf16 v8bf;
typedef __attribute__((ext_vector_type(4)))  __bf16 v4bf;
typedef __attribute__((ext_vector_type(8)))  float  v8f;
typedef __attribute__((ext_vector_type(4)))  float  v4f;
typedef __attribute__((ext_vector_type(4)))  unsigned int u32x4;
typedef __attribute__((ext_vector_type(4)))  int    i32x4;
typedef __attribute__((ext_vector_type(8)))  int    i32x8;

#define TILE_M 128
#define TILE_N 64
#define TILE_K 32
#define LDSTR  48   // padded LDS row stride in bf16 (96 B, 16B-aligned rows)

#if __has_builtin(__builtin_amdgcn_tensor_load_to_lds) && \
    __has_builtin(__builtin_amdgcn_s_wait_tensorcnt)
#define HAVE_TDM 1
#else
#define HAVE_TDM 0
#endif

__device__ __forceinline__ __bf16 f2bf(float f) {
    unsigned u = __builtin_bit_cast(unsigned, f);
    unsigned r = u + 0x7FFFu + ((u >> 16) & 1u);      // round-to-nearest-even
    unsigned short h = (unsigned short)(r >> 16);
    return __builtin_bit_cast(__bf16, h);
}

// fast sigmoid: v_exp_f32 + v_rcp_f32 (no IEEE div chain)
__device__ __forceinline__ float sigm(float x) {
    return __builtin_amdgcn_rcpf(1.0f + __expf(-x));
}

// ---------------------------------------------------------------------------
// one-time fp32 -> bf16 conversion (vectorized x4)
// ---------------------------------------------------------------------------
__global__ __launch_bounds__(256)
void cvt_f32_to_bf16(const float* __restrict__ in, __bf16* __restrict__ out, int n4) {
    const int i = blockIdx.x * 256 + threadIdx.x;
    if (i >= n4) return;
    const v4f x = ((const v4f*)in)[i];
    v4bf y;
    y[0] = f2bf(x[0]); y[1] = f2bf(x[1]); y[2] = f2bf(x[2]); y[3] = f2bf(x[3]);
    ((v4bf*)out)[i] = y;
}

// ---------------------------------------------------------------------------
// one-time transpose-convert: in [K x N] fp32 -> out [N x K] bf16.
// 64x64 tiles through LDS; both global accesses coalesced.
// ---------------------------------------------------------------------------
__global__ __launch_bounds__(256)
void cvt_transpose_bf16(const float* __restrict__ in, __bf16* __restrict__ out,
                        int K, int N) {
    __shared__ __bf16 tile[64][65];
    const int bk = blockIdx.x * 64;
    const int bn = blockIdx.y * 64;
    const int tx = threadIdx.x & 63;
    const int ty = threadIdx.x >> 6;          // 0..3
    #pragma unroll
    for (int r = 0; r < 16; ++r) {
        const int k = ty * 16 + r;
        tile[k][tx] = f2bf(in[(size_t)(bk + k) * N + (bn + tx)]);
    }
    __syncthreads();
    #pragma unroll
    for (int r = 0; r < 16; ++r) {
        const int n = ty * 16 + r;
        out[(size_t)(bn + n) * K + (bk + tx)] = tile[tx][n];
    }
}

// ---------------------------------------------------------------------------
// Y = X[MxK](bf16) @ Wt^T, where Wt is W transposed: [N x K](bf16).
// fp32 accum via v_wmma_f32_16x16x32_bf16.
// MODE 0: Y0 = swish(x) * 128^-0.5               (q path)
// MODE 1: Y0 = 1 - sigmoid(x); Y1 = sigmoid(x)   (k and exp(g) from f path)
// MODE 2: Y0 = x                                  (v path / output proj)
// Block 256 thr (8 waves); block tile 128x64; wave tile 32x32 (2x2 WMMA).
// Both tiles are contiguous row-copies -> staged by TDM, double-buffered.
// ---------------------------------------------------------------------------
template <int MODE>
__global__ __launch_bounds__(256)
void hgrn2_gemm_wmma(const __bf16* __restrict__ X, const __bf16* __restrict__ Wt,
                     float* __restrict__ Y0, float* __restrict__ Y1,
                     int M, int N, int K) {
    __shared__ __bf16 ldsA[2][TILE_M * LDSTR];
    __shared__ __bf16 ldsB[2][TILE_N * LDSTR];

    const int tid  = threadIdx.x;
    const int lane = tid & 31;
    const int wave = tid >> 5;        // 0..7
    const int wm   = wave >> 1;       // 0..3 -> 32-row strip
    const int wn   = wave & 1;        // 0..1 -> 32-col strip

    const int blockM = blockIdx.x * TILE_M;
    const int blockN = blockIdx.y * TILE_N;

    const int lhalf = (lane >> 4) & 1;
    const int lmod  = lane & 15;
    const int kb0   = lhalf * 8;
    const int kb1   = 16 + lhalf * 8;

    v8f acc[2][2] = {};
    const int niter = K / TILE_K;

#if HAVE_TDM
    // TDM descriptor issue: 2D tile 32(K,x) x rows(y), 2B elements,
    // LDS pad: after every 16 DWORDs (64B row) insert 8 DWORDs -> 48-elem stride.
    auto tdm_issue = [&](const __bf16* gptr, unsigned lds_base, int rows, int trows) {
        const unsigned long long ga = (unsigned long long)(size_t)gptr;
        u32x4 g0;
        g0[0] = 1u;                                        // count=1, user mode
        g0[1] = lds_base;                                  // lds_addr (bytes)
        g0[2] = (unsigned)(ga & 0xFFFFFFFFu);              // global_addr[31:0]
        g0[3] = (unsigned)((ga >> 32) & 0x01FFFFFFu)       // global_addr[56:32]
              | (2u << 30);                                // type = 2 ("image")
        i32x8 g1;
        g1[0] = (int)((1u << 16)       // data_size = 1 -> 2 bytes
              | (1u << 20)             // pad_enable
              | (3u << 22)             // pad_interval code 3 -> 16 DWORDs
              | (7u << 25));           // pad_amount   code 7 -> 8 DWORDs
        g1[1] = (int)((unsigned)K << 16);                  // tensor_dim0 = K
        g1[2] = (int)(((unsigned)trows & 0xFFFFu) << 16);  // tensor_dim1
        g1[3] = (int)(32u << 16);                          // tile_dim0 = 32 (K)
        g1[4] = rows;                                      // tile_dim1
        g1[5] = (int)K;                                    // tensor_dim0_stride
        g1[6] = 0;
        g1[7] = 0;
        i32x4 z4 = {0, 0, 0, 0};
#if __clang_major__ >= 23
        i32x8 z8 = {0, 0, 0, 0, 0, 0, 0, 0};
        __builtin_amdgcn_tensor_load_to_lds(g0, g1, z4, z4, z8, 0);
#else
        __builtin_amdgcn_tensor_load_to_lds(g0, g1, z4, z4, 0);
#endif
    };
    auto issue_pair = [&](int it, int buf) {
        const int k0 = it * TILE_K;
        tdm_issue(X  + (size_t)blockM * K + k0,
                  (unsigned)(size_t)(void*)&ldsA[buf][0], TILE_M, M);
        tdm_issue(Wt + (size_t)blockN * K + k0,
                  (unsigned)(size_t)(void*)&ldsB[buf][0], TILE_N, N);
    };
    if (wave == 0) issue_pair(0, 0);
#endif

    for (int i = 0; i < niter; ++i) {
        const int cur = i & 1;
#if HAVE_TDM
        if (wave == 0) {
            if (i + 1 < niter) {
                issue_pair(i + 1, cur ^ 1);            // buffer freed at end of i-1
                __builtin_amdgcn_s_wait_tensorcnt(2);  // in-order: pair(i) complete
            } else {
                __builtin_amdgcn_s_wait_tensorcnt(0);
            }
        }
        __syncthreads();                               // LDS tiles visible to all waves
#else
        const int k0 = i * TILE_K;
        {   // A tile: straight row copy, 2x 16B per thread
            const int row = tid >> 1;
            const int cb  = (tid & 1) * 16;
            const v8bf* src = (const v8bf*)(X + (size_t)(blockM + row) * K + (k0 + cb));
            v8bf* dst = (v8bf*)&ldsA[cur][row * LDSTR + cb];
            dst[0] = src[0];
            dst[1] = src[1];
        }
        {   // B tile: straight row copy from Wt, 16B per thread
            const int row = tid >> 2;                 // 0..63
            const int cb  = (tid & 3) * 8;            // 0,8,16,24
            *(v8bf*)&ldsB[cur][row * LDSTR + cb] =
                *(const v8bf*)(Wt + (size_t)(blockN + row) * K + (k0 + cb));
        }
        __syncthreads();
#endif
        // ---- fragments + 4 WMMAs -----------------------------------------
        v16bf a[2], b[2];
        #pragma unroll
        for (int mi = 0; mi < 2; ++mi) {
            const __bf16* ar = &ldsA[cur][(wm * 32 + mi * 16 + lmod) * LDSTR];
            v8bf alo = *(const v8bf*)(ar + kb0);
            v8bf ahi = *(const v8bf*)(ar + kb1);
            a[mi] = __builtin_shufflevector(alo, ahi, 0,1,2,3,4,5,6,7,8,9,10,11,12,13,14,15);
        }
        #pragma unroll
        for (int ni = 0; ni < 2; ++ni) {
            const __bf16* br = &ldsB[cur][(wn * 32 + ni * 16 + lmod) * LDSTR];
            v8bf blo = *(const v8bf*)(br + kb0);
            v8bf bhi = *(const v8bf*)(br + kb1);
            b[ni] = __builtin_shufflevector(blo, bhi, 0,1,2,3,4,5,6,7,8,9,10,11,12,13,14,15);
        }
        #pragma unroll
        for (int mi = 0; mi < 2; ++mi)
            #pragma unroll
            for (int ni = 0; ni < 2; ++ni)
                acc[mi][ni] = __builtin_amdgcn_wmma_f32_16x16x32_bf16(
                    false, a[mi], false, b[ni], (short)0, acc[mi][ni], false, false);
        __syncthreads();                               // tile consumed -> buffer reusable
    }

    // ---- epilogue (hoisted row pointers, stride-N stores) -----------------
    #pragma unroll
    for (int mi = 0; mi < 2; ++mi) {
        #pragma unroll
        for (int ni = 0; ni < 2; ++ni) {
            const int gn  = blockN + wn * 32 + ni * 16 + lmod;
            const int gm0 = blockM + wm * 32 + mi * 16 + lhalf * 8;
            float* __restrict__ p0 = Y0 + (size_t)gm0 * N + gn;
            float* __restrict__ p1 = (MODE == 1) ? (Y1 + (size_t)gm0 * N + gn) : nullptr;
            #pragma unroll
            for (int r = 0; r < 8; ++r) {
                const float x = acc[mi][ni][r];
                const size_t off = (size_t)r * N;
                if (MODE == 0) {
                    p0[off] = x * sigm(x) * 0.08838834764831845f;  // swish * 128^-0.5
                } else if (MODE == 1) {
                    const float s = sigm(x);
                    p0[off] = 1.0f - s;   // k = 1 - sigmoid(f)
                    p1[off] = s;          // exp(g) = exp(logsigmoid(f)) = sigmoid(f)
                } else {
                    p0[off] = x;
                }
            }
        }
    }
}

// ---------------------------------------------------------------------------
// GLA scan: one 512-thread WG per (b,h). State S[128x128] in registers:
// thread owns column j = tid&127, rows [ib, ib+32). Row-indexed q/k/eg LDS
// reads are wave-broadcasts; o_t reduced 4-way through LDS.
// ---------------------------------------------------------------------------
__global__ __launch_bounds__(512)
void hgrn2_gla_scan(const float* __restrict__ q, const float* __restrict__ k,
                    const float* __restrict__ eg, const float* __restrict__ v,
                    float* __restrict__ o, int T) {
    __shared__ float lq[128], lk[128], leg[128], lv[128];
    __shared__ float lo[4][128];

    const int bh  = blockIdx.x;
    const int b   = bh >> 3;
    const int h   = bh & 7;
    const int tid = threadIdx.x;
    const int j   = tid & 127;
    const int grp = tid >> 7;
    const int ib  = grp << 5;

    float S[32];
    #pragma unroll
    for (int r = 0; r < 32; ++r) S[r] = 0.0f;

    float* stage = (grp == 0) ? lq : (grp == 1) ? lk : (grp == 2) ? leg : lv;
    const float* src = (grp == 0) ? q : (grp == 1) ? k : (grp == 2) ? eg : v;

    for (int t = 0; t < T; ++t) {
        const size_t base = ((size_t)(b * T + t) << 10) + (h << 7);
        stage[j] = src[base + j];
        __builtin_prefetch(src + base + 1024 + j, 0, 0);   // next timestep line
        __syncthreads();

        const float vj = lv[j];
        float acc = 0.0f;
        #pragma unroll
        for (int r = 0; r < 32; ++r) {
            const float s = leg[ib + r] * S[r] + lk[ib + r] * vj;
            S[r] = s;
            acc = fmaf(lq[ib + r], s, acc);
        }
        lo[grp][j] = acc;
        __syncthreads();

        if (tid < 128)
            o[base + tid] = lo[0][tid] + lo[1][tid] + lo[2][tid] + lo[3][tid];
    }
}

// ---------------------------------------------------------------------------
// RMSNorm (weight, no bias) -> bf16 output for the final WMMA GEMM
// ---------------------------------------------------------------------------
__global__ __launch_bounds__(256)
void hgrn2_rmsnorm(const float* __restrict__ o, const float* __restrict__ gw,
                   __bf16* __restrict__ out, int D) {
    __shared__ float red[256];
    const int row = blockIdx.x;
    const float* src = o + (size_t)row * D;

    float ss = 0.0f;
    for (int d = threadIdx.x; d < D; d += 256) { const float x = src[d]; ss = fmaf(x, x, ss); }
    red[threadIdx.x] = ss;
    __syncthreads();
    for (int w = 128; w > 0; w >>= 1) {
        if (threadIdx.x < w) red[threadIdx.x] += red[threadIdx.x + w];
        __syncthreads();
    }
    const float r = rsqrtf(red[0] / (float)D + 1e-5f);
    for (int d = threadIdx.x; d < D; d += 256)
        out[(size_t)row * D + d] = f2bf(src[d] * r * gw[d]);
}

// ---------------------------------------------------------------------------
extern "C" void kernel_launch(void* const* d_in, const int* in_sizes, int n_in,
                              void* d_out, int out_size, void* d_ws, size_t ws_size,
                              hipStream_t stream) {
    const float* hs = (const float*)d_in[0];
    const float* Wq = (const float*)d_in[1];
    const float* Wf = (const float*)d_in[2];
    const float* Wi = (const float*)d_in[3];
    const float* gw = (const float*)d_in[4];
    const float* Wo = (const float*)d_in[5];

    const int B = 4, T = 2048, D = 1024, H = 8;
    const int M = B * T;                       // 8192
    const size_t PLANE = (size_t)M * D;        // 8M elements

    char* wsp = (char*)d_ws;
    __bf16* hs_bf = (__bf16*)wsp;  wsp += PLANE * 2;            // 16 MB
    __bf16* Wq_bf = (__bf16*)wsp;  wsp += (size_t)D * D * 2;    //  2 MB each (transposed)
    __bf16* Wf_bf = (__bf16*)wsp;  wsp += (size_t)D * D * 2;
    __bf16* Wi_bf = (__bf16*)wsp;  wsp += (size_t)D * D * 2;
    __bf16* Wo_bf = (__bf16*)wsp;  wsp += (size_t)D * D * 2;
    float* q   = (float*)wsp;      wsp += PLANE * 4;            // 32 MB each
    float* kk  = (float*)wsp;      wsp += PLANE * 4;
    float* eg  = (float*)wsp;      wsp += PLANE * 4;            // exp(g) = sigmoid(f)
    float* v   = (float*)wsp;      wsp += PLANE * 4;
    float* o   = (float*)wsp;      wsp += PLANE * 4;
    __bf16* onm_bf = hs_bf;        // hs_bf dead after projections -> reuse

    // one-time conversions: hs flat, weights transposed to [N x K] bf16
    const int n4hs = (int)(PLANE / 4);
    cvt_f32_to_bf16<<<dim3((n4hs + 255) / 256), dim3(256), 0, stream>>>(hs, hs_bf, n4hs);
    dim3 tgrid(D / 64, D / 64), tblk(256);
    cvt_transpose_bf16<<<tgrid, tblk, 0, stream>>>(Wq, Wq_bf, D, D);
    cvt_transpose_bf16<<<tgrid, tblk, 0, stream>>>(Wf, Wf_bf, D, D);
    cvt_transpose_bf16<<<tgrid, tblk, 0, stream>>>(Wi, Wi_bf, D, D);
    cvt_transpose_bf16<<<tgrid, tblk, 0, stream>>>(Wo, Wo_bf, D, D);

    dim3 ggrid(M / TILE_M, D / TILE_N);        // 64 x 16
    dim3 gblk(256);

    hgrn2_gemm_wmma<0><<<ggrid, gblk, 0, stream>>>(hs_bf, Wq_bf, q,  nullptr, M, D, D);
    hgrn2_gemm_wmma<1><<<ggrid, gblk, 0, stream>>>(hs_bf, Wf_bf, kk, eg,      M, D, D);
    hgrn2_gemm_wmma<2><<<ggrid, gblk, 0, stream>>>(hs_bf, Wi_bf, v,  nullptr, M, D, D);

    hgrn2_gla_scan<<<dim3(B * H), dim3(512), 0, stream>>>(q, kk, eg, v, o, T);

    hgrn2_rmsnorm<<<dim3(M), dim3(256), 0, stream>>>(o, gw, onm_bf, D);

    hgrn2_gemm_wmma<2><<<ggrid, gblk, 0, stream>>>(onm_bf, Wo_bf, (float*)d_out, nullptr, M, D, D);
}
